// TransformerBlock_3478923509982
// MI455X (gfx1250) — matrix operand order repaired and checked by reading, not compiled, not run on hardware
//
#include <hip/hip_runtime.h>
#include <hip/hip_bf16.h>
#include <math.h>

// ---------------------------------------------------------------------------
// Problem constants (B,T,C,H from the reference)
// ---------------------------------------------------------------------------
#define Bz 4
#define Tz 1024
#define Cz 1024
#define Hz 16
#define Dz 64
#define Fz 4096
#define BTz (Bz * Tz)

typedef unsigned short u16;
typedef __bf16  v8bf  __attribute__((ext_vector_type(8)));
typedef __bf16  v16bf __attribute__((ext_vector_type(16)));
typedef float   v8f   __attribute__((ext_vector_type(8)));

static __device__ __forceinline__ u16 f2b(float f) {
    unsigned int u = __float_as_uint(f);
    unsigned int r = u + 0x7FFFu + ((u >> 16) & 1u);   // round-to-nearest-even
    return (u16)(r >> 16);
}

static __device__ __forceinline__ v8f vzero8() {
    v8f z = {0.f, 0.f, 0.f, 0.f, 0.f, 0.f, 0.f, 0.f};
    return z;
}

static __device__ __forceinline__ v16bf pack16(v8bf lo, v8bf hi) {
    return __builtin_shufflevector(lo, hi, 0, 1, 2, 3, 4, 5, 6, 7,
                                            8, 9, 10, 11, 12, 13, 14, 15);
}

static __device__ __forceinline__ v8f wmma_bf16(v16bf a, v16bf b, v8f c) {
    return __builtin_amdgcn_wmma_f32_16x16x32_bf16(
        /*neg_a=*/false, a, /*neg_b=*/false, b,
        /*c_mod=*/(short)0, c, /*reuse_a=*/false, /*reuse_b=*/false);
}

// CDNA5 async global->LDS copy, 16B per lane, tracked by ASYNCcnt.
static __device__ __forceinline__ void async_copy_b128(const u16* gsrc, u16* lds_dst) {
    unsigned int       la = (unsigned int)(size_t)lds_dst;       // LDS byte addr
    unsigned long long ga = (unsigned long long)(size_t)gsrc;    // global addr
    asm volatile("global_load_async_to_lds_b128 %0, %1, off"
                 :: "v"(la), "v"(ga) : "memory");
}
static __device__ __forceinline__ void wait_async_le8() {
    asm volatile("s_wait_asynccnt 8" ::: "memory");
}
static __device__ __forceinline__ void wait_async_0() {
    asm volatile("s_wait_asynccnt 0" ::: "memory");
}

// ---------------------------------------------------------------------------
// LayerNorm (f32 in) -> bf16 out.  One block per row of C=1024, 256 threads.
// ---------------------------------------------------------------------------
__global__ __launch_bounds__(256) void ln_bf16_kernel(
    const float* __restrict__ x, const float* __restrict__ g,
    const float* __restrict__ b, u16* __restrict__ out) {
    const int row = blockIdx.x;
    const int tid = threadIdx.x;
    const float* xr = x + (size_t)row * Cz;

    float v[4];
    float s = 0.f;
#pragma unroll
    for (int i = 0; i < 4; ++i) { v[i] = xr[tid + i * 256]; s += v[i]; }

    __shared__ float red[256];
    red[tid] = s;
#pragma unroll
    for (int off = 128; off > 0; off >>= 1) {
        __syncthreads();
        if (tid < off) red[tid] += red[tid + off];
    }
    __syncthreads();
    const float mean = red[0] * (1.0f / Cz);
    __syncthreads();

    float q = 0.f;
#pragma unroll
    for (int i = 0; i < 4; ++i) { float d = v[i] - mean; q += d * d; }
    red[tid] = q;
#pragma unroll
    for (int off = 128; off > 0; off >>= 1) {
        __syncthreads();
        if (tid < off) red[tid] += red[tid + off];
    }
    __syncthreads();
    const float rstd = rsqrtf(red[0] * (1.0f / Cz) + 1e-5f);

#pragma unroll
    for (int i = 0; i < 4; ++i) {
        int c = tid + i * 256;
        out[(size_t)row * Cz + c] = f2b((v[i] - mean) * rstd * g[c] + b[c]);
    }
}

// ---------------------------------------------------------------------------
// Weight conversion: build bf16, transposed [N][K] copies.
//   dst[n*K + c] = src[c*ldSrc + n]
// ---------------------------------------------------------------------------
__global__ __launch_bounds__(256) void transpose_bf16_kernel(
    const float* __restrict__ src, u16* __restrict__ dst,
    int N, int K, int ldSrc) {
    int idx = blockIdx.x * 256 + threadIdx.x;
    if (idx >= N * K) return;
    int n = idx / K, c = idx % K;
    dst[idx] = f2b(src[(size_t)c * ldSrc + n]);
}

// WqkvT[n][c]: n = proj*C + h*D + d ; src Wq/Wk/Wv[h][c][d]
__global__ __launch_bounds__(256) void wqkv_convert_kernel(
    const float* __restrict__ Wq, const float* __restrict__ Wk,
    const float* __restrict__ Wv, u16* __restrict__ dst) {
    int idx = blockIdx.x * 256 + threadIdx.x;       // over 3072*1024
    int n = idx >> 10;                               // 0..3071
    int c = idx & 1023;
    int proj = n >> 10;
    int rest = n & 1023;
    int h = rest >> 6;
    int d = rest & 63;
    const float* W = (proj == 0) ? Wq : (proj == 1) ? Wk : Wv;
    dst[idx] = f2b(W[((size_t)h * Cz + c) * Dz + d]);
}

// ---------------------------------------------------------------------------
// Generic bf16 GEMM:  C[M,N] = A[M,K] * Bt[N,K]^T (+bias)(+relu)
// block = 256 thr = 8 waves; tile 128x128x64; wave tile 64x32 (4x2 subtiles).
// Double-buffered LDS fed by async global->LDS copies (ASYNCcnt pipelined).
// ---------------------------------------------------------------------------
#define GEMM_RELU 1
#define GEMM_QKV  2

__global__ __launch_bounds__(256) void gemm_bf16_kernel(
    const u16* __restrict__ A, int lda,
    const u16* __restrict__ Bt, int ldb,
    int K,
    const float* __restrict__ bias,
    float* __restrict__ outF, u16* __restrict__ outH, int ldo,
    int mode,
    u16* __restrict__ qOut, u16* __restrict__ kOut, u16* __restrict__ vOut) {
    __shared__ alignas(64) u16 As[2][128][64];
    __shared__ alignas(64) u16 Bs[2][128][64];

    const int tid    = threadIdx.x;
    const int wid    = tid >> 5;
    const int lane   = tid & 31;
    const int lane15 = lane & 15;
    const int hi     = lane >> 4;
    const int waveM  = wid >> 2;        // 0..1
    const int waveN  = wid & 3;         // 0..3
    const int m0     = blockIdx.y * 128;
    const int n0     = blockIdx.x * 128;

    v8f acc[4][2];
#pragma unroll
    for (int mi = 0; mi < 4; ++mi)
#pragma unroll
        for (int ni = 0; ni < 2; ++ni) acc[mi][ni] = vzero8();

    // each thread owns 4 A-chunks + 4 B-chunks (16B each) of a 128x64 tile
    auto issue_tile = [&](int buf, int k0) {
#pragma unroll
        for (int i = 0; i < 4; ++i) {
            int c  = tid + i * 256;            // 0..1023
            int r  = c >> 3;
            int cc = (c & 7) * 8;
            async_copy_b128(A  + (size_t)(m0 + r) * lda + k0 + cc, &As[buf][r][cc]);
            async_copy_b128(Bt + (size_t)(n0 + r) * ldb + k0 + cc, &Bs[buf][r][cc]);
        }
    };

    const int ktiles = K >> 6;
    issue_tile(0, 0);

    for (int kt = 0; kt < ktiles; ++kt) {
        const int cur = kt & 1;
        const bool hasNext = (kt + 1) < ktiles;
        if (hasNext) {
            issue_tile(cur ^ 1, (kt + 1) << 6);   // overlap next copy w/ compute
            wait_async_le8();                      // current tile's 8 copies done
        } else {
            wait_async_0();
        }
        __syncthreads();

#pragma unroll
        for (int kk = 0; kk < 64; kk += 32) {
            v16bf aF[4];
#pragma unroll
            for (int mi = 0; mi < 4; ++mi) {
                const u16* arow = &As[cur][waveM * 64 + mi * 16 + lane15][0];
                int c0 = kk + hi * 16;
                v8bf lo = *(const v8bf*)(arow + c0);
                v8bf hh = *(const v8bf*)(arow + c0 + 8);
                aF[mi] = pack16(lo, hh);
            }
            v16bf bF[2];
#pragma unroll
            for (int ni = 0; ni < 2; ++ni)
                bF[ni] = *(const v16bf*)(&Bs[cur][waveN * 32 + ni * 16 + lane15][kk + hi * 16]);
#pragma unroll
            for (int mi = 0; mi < 4; ++mi)
#pragma unroll
                for (int ni = 0; ni < 2; ++ni)
                    acc[mi][ni] = wmma_bf16(aF[mi], bF[ni], acc[mi][ni]);
        }
        __syncthreads();   // everyone done reading before buffer is re-filled
    }

    // epilogue
#pragma unroll
    for (int mi = 0; mi < 4; ++mi) {
#pragma unroll
        for (int ni = 0; ni < 2; ++ni) {
            const int col = n0 + waveN * 32 + ni * 16 + lane15;
            const float bv = bias ? bias[col] : 0.f;
#pragma unroll
            for (int r = 0; r < 8; ++r) {
                const int row = m0 + waveM * 64 + mi * 16 + hi * 8 + r;
                float v = acc[mi][ni][r] + bv;
                if (mode & GEMM_RELU) v = fmaxf(v, 0.f);
                if (mode & GEMM_QKV) {
                    const int proj = col >> 10;
                    const int rest = col & 1023;
                    const int h = rest >> 6;
                    const int d = rest & 63;
                    const int b = row >> 10;
                    const int t = row & 1023;
                    const u16 bvv = f2b(v);
                    if (proj == 0)
                        qOut[(((size_t)b * Hz + h) * Tz + t) * Dz + d] = bvv;
                    else if (proj == 1)
                        kOut[(((size_t)b * Hz + h) * Tz + t) * Dz + d] = bvv;
                    else
                        vOut[(((size_t)b * Hz + h) * Dz + d) * Tz + t] = bvv;  // transposed
                } else if (outF) {
                    outF[(size_t)row * ldo + col] = v;
                } else {
                    outH[(size_t)row * ldo + col] = f2b(v);
                }
            }
        }
    }
}

// ---------------------------------------------------------------------------
// Attention: per (b,h), flash-style over s-tiles of 64. Block = 128 thr = 4
// waves; wave handles 16 query rows. q@k^T and P@V via bf16 WMMA.
// K/V tiles staged with async global->LDS copies.
// ---------------------------------------------------------------------------
__global__ __launch_bounds__(128) void attn_kernel(
    const u16* __restrict__ qb, const u16* __restrict__ kb,
    const u16* __restrict__ vTb, const unsigned char* __restrict__ pad,
    u16* __restrict__ attout) {
    __shared__ alignas(64) u16 Kt[64][64];
    __shared__ alignas(64) u16 Vt[64][64];   // Vt[d][s]
    __shared__ alignas(64) u16 Pt[4][16][64];
    __shared__ float pads[64];

    const int bh = blockIdx.y;
    const int b  = bh / Hz;
    const int h  = bh % Hz;
    const int tblk = blockIdx.x;                 // 0..T/64-1
    const int tid  = threadIdx.x;
    const int wave = tid >> 5;
    const int lane = tid & 31;
    const int lane15 = lane & 15;
    const int hi = lane >> 4;
    const int tbase = tblk * 64 + wave * 16;

    const u16* qhead = qb  + ((size_t)bh * Tz) * Dz;
    const u16* khead = kb  + ((size_t)bh * Tz) * Dz;
    const u16* vhead = vTb + ((size_t)bh * Dz) * Tz;

    // q fragments (A layout, 16x32 per k-step, D=64 -> 2 fragments), stationary
    v16bf qf[2];
    {
        const u16* qr = qhead + (size_t)(tbase + lane15) * Dz;
#pragma unroll
        for (int kkf = 0; kkf < 2; ++kkf) {
            int c0 = kkf * 32 + hi * 16;
            qf[kkf] = pack16(*(const v8bf*)(qr + c0), *(const v8bf*)(qr + c0 + 8));
        }
    }

    float mrow[8], lrow[8];
    v8f O[4];
#pragma unroll
    for (int r = 0; r < 8; ++r) { mrow[r] = -3e38f; lrow[r] = 0.f; }
#pragma unroll
    for (int n = 0; n < 4; ++n) O[n] = vzero8();

    const int ntiles = tblk + 1;                 // causal: s-tiles <= t-tile
    for (int jt = 0; jt < ntiles; ++jt) {
        const int sbase = jt * 64;
        // async stage of K (row-major [s][d]) and V^T (row [d], cols s)
#pragma unroll
        for (int i = 0; i < 4; ++i) {
            int c  = tid + i * 128;              // 0..511
            int r  = c >> 3;
            int cc = (c & 7) * 8;
            async_copy_b128(khead + (size_t)(sbase + r) * Dz + cc, &Kt[r][cc]);
            async_copy_b128(vhead + (size_t)r * Tz + sbase + cc,  &Vt[r][cc]);
        }
        if (tid < 64) pads[tid] = pad[(size_t)b * Tz + sbase + tid] ? 1.f : 0.f;
        wait_async_0();
        __syncthreads();

        // S = q @ k^T  (16 x 64)
        v8f S[4];
#pragma unroll
        for (int n = 0; n < 4; ++n) {
            v16bf bk0 = *(const v16bf*)(&Kt[n * 16 + lane15][hi * 16]);
            v16bf bk1 = *(const v16bf*)(&Kt[n * 16 + lane15][32 + hi * 16]);
            v8f a = vzero8();
            a = wmma_bf16(qf[0], bk0, a);
            a = wmma_bf16(qf[1], bk1, a);
            S[n] = a;
        }

        // scale + causal + padding masks, rowwise online softmax
        float tmax[8];
#pragma unroll
        for (int r = 0; r < 8; ++r) {
            const int t = tbase + hi * 8 + r;
            float mx = -3e38f;
#pragma unroll
            for (int n = 0; n < 4; ++n) {
                const int s = sbase + n * 16 + lane15;
                float val = S[n][r] * 0.03125f;          // C^-0.5 = 1/32
                if (s > t) val = -3e38f;                 // causal -inf
                if (pads[n * 16 + lane15] != 0.f) val = -1e9f;  // pad override
                S[n][r] = val;
                mx = fmaxf(mx, val);
            }
            tmax[r] = mx;
        }
#pragma unroll
        for (int mk = 1; mk < 16; mk <<= 1)
#pragma unroll
            for (int r = 0; r < 8; ++r)
                tmax[r] = fmaxf(tmax[r], __shfl_xor(tmax[r], mk, 32));

        float alpha[8];
#pragma unroll
        for (int r = 0; r < 8; ++r) {
            float mn = fmaxf(mrow[r], tmax[r]);
            alpha[r] = __expf(mrow[r] - mn);
            mrow[r] = mn;
        }
        float psum[8];
#pragma unroll
        for (int r = 0; r < 8; ++r) {
            float s = 0.f;
#pragma unroll
            for (int n = 0; n < 4; ++n) {
                float p = __expf(S[n][r] - mrow[r]);
                S[n][r] = p;
                s += p;
            }
            psum[r] = s;
        }
#pragma unroll
        for (int mk = 1; mk < 16; mk <<= 1)
#pragma unroll
            for (int r = 0; r < 8; ++r) psum[r] += __shfl_xor(psum[r], mk, 32);
#pragma unroll
        for (int r = 0; r < 8; ++r) lrow[r] = lrow[r] * alpha[r] + psum[r];
#pragma unroll
        for (int n = 0; n < 4; ++n)
#pragma unroll
            for (int r = 0; r < 8; ++r) O[n][r] *= alpha[r];

        // P (D-layout) -> LDS -> A-fragment layout
#pragma unroll
        for (int n = 0; n < 4; ++n)
#pragma unroll
            for (int r = 0; r < 8; ++r)
                Pt[wave][hi * 8 + r][n * 16 + lane15] = f2b(S[n][r]);
        __syncthreads();

        v16bf pf[2];
#pragma unroll
        for (int kkf = 0; kkf < 2; ++kkf) {
            int c0 = kkf * 32 + hi * 16;
            const u16* pr = &Pt[wave][lane15][0];
            pf[kkf] = pack16(*(const v8bf*)(pr + c0), *(const v8bf*)(pr + c0 + 8));
        }
#pragma unroll
        for (int n = 0; n < 4; ++n) {
            v16bf bv0 = *(const v16bf*)(&Vt[n * 16 + lane15][hi * 16]);
            v16bf bv1 = *(const v16bf*)(&Vt[n * 16 + lane15][32 + hi * 16]);
            O[n] = wmma_bf16(pf[0], bv0, O[n]);
            O[n] = wmma_bf16(pf[1], bv1, O[n]);
        }
        __syncthreads();
    }

    // normalize + store bf16 into concat-heads layout [B*T][C]
#pragma unroll
    for (int n = 0; n < 4; ++n) {
#pragma unroll
        for (int r = 0; r < 8; ++r) {
            const int t = tbase + hi * 8 + r;
            const int d = n * 16 + lane15;
            attout[((size_t)b * Tz + t) * Cz + h * Dz + d] =
                f2b(O[n][r] * (1.f / lrow[r]));
        }
    }
}

// ---------------------------------------------------------------------------
// Host launcher
// ---------------------------------------------------------------------------
extern "C" void kernel_launch(void* const* d_in, const int* in_sizes, int n_in,
                              void* d_out, int out_size, void* d_ws, size_t ws_size,
                              hipStream_t stream) {
    const float* x     = (const float*)d_in[0];
    const unsigned char* pmask = (const unsigned char*)d_in[1];  // jax bool
    const float* Wq    = (const float*)d_in[2];
    const float* Wk    = (const float*)d_in[3];
    const float* Wv    = (const float*)d_in[4];
    const float* Wo    = (const float*)d_in[5];
    const float* bo    = (const float*)d_in[6];
    const float* ln1_g = (const float*)d_in[7];
    const float* ln1_b = (const float*)d_in[8];
    const float* ln2_g = (const float*)d_in[9];
    const float* ln2_b = (const float*)d_in[10];
    const float* W1    = (const float*)d_in[11];
    const float* b1    = (const float*)d_in[12];
    const float* W2    = (const float*)d_in[13];
    const float* b2    = (const float*)d_in[14];
    float* out = (float*)d_out;

    char* ws = (char*)d_ws;
    size_t off = 0;
    auto take = [&](size_t bytes) -> void* {
        void* p = ws + off;
        off += (bytes + 255) & ~(size_t)255;
        return p;
    };
    u16*   hb     = (u16*)take((size_t)BTz * Cz * 2);        // LN1 out, bf16
    u16*   WqkvT  = (u16*)take((size_t)3 * Cz * Cz * 2);     // [3C][C]
    u16*   WoT    = (u16*)take((size_t)Cz * Cz * 2);
    u16*   W1T    = (u16*)take((size_t)Fz * Cz * 2);         // [F][C]
    u16*   W2T    = (u16*)take((size_t)Cz * Fz * 2);         // [C][F]
    u16*   qb     = (u16*)take((size_t)Bz * Hz * Tz * Dz * 2);
    u16*   kbuf   = (u16*)take((size_t)Bz * Hz * Tz * Dz * 2);
    u16*   vTb    = (u16*)take((size_t)Bz * Hz * Dz * Tz * 2);
    u16*   attoutb= (u16*)take((size_t)BTz * Cz * 2);
    float* out1   = (float*)take((size_t)BTz * Cz * 4);      // proj out, f32
    u16*   h2b    = (u16*)take((size_t)BTz * Cz * 2);        // LN2 out
    u16*   a1b    = (u16*)take((size_t)BTz * Fz * 2);        // relu(ffn1), bf16

    // weight conversion (independent)
    wqkv_convert_kernel<<<(3 * Cz * Cz) / 256, 256, 0, stream>>>(Wq, Wk, Wv, WqkvT);
    transpose_bf16_kernel<<<(Cz * Cz) / 256, 256, 0, stream>>>(Wo, WoT, Cz, Cz, Cz);
    transpose_bf16_kernel<<<(Fz * Cz) / 256, 256, 0, stream>>>(W1, W1T, Fz, Cz, Fz);
    transpose_bf16_kernel<<<(Cz * Fz) / 256, 256, 0, stream>>>(W2, W2T, Cz, Fz, Cz);

    // LN1
    ln_bf16_kernel<<<BTz, 256, 0, stream>>>(x, ln1_g, ln1_b, hb);

    // fused QKV projection: M=4096, N=3072, K=1024, scatter epilogue
    gemm_bf16_kernel<<<dim3(3 * Cz / 128, BTz / 128), 256, 0, stream>>>(
        hb, Cz, WqkvT, Cz, Cz, nullptr, nullptr, nullptr, 0,
        GEMM_QKV, qb, kbuf, vTb);

    // attention
    attn_kernel<<<dim3(Tz / 64, Bz * Hz), 128, 0, stream>>>(qb, kbuf, vTb, pmask, attoutb);

    // output projection: M=4096, N=1024, K=1024 -> f32
    gemm_bf16_kernel<<<dim3(Cz / 128, BTz / 128), 256, 0, stream>>>(
        attoutb, Cz, WoT, Cz, Cz, bo, out1, nullptr, Cz, 0,
        nullptr, nullptr, nullptr);

    // LN2
    ln_bf16_kernel<<<BTz, 256, 0, stream>>>(out1, ln2_g, ln2_b, h2b);

    // FFN1: M=4096, N=4096, K=1024, relu -> bf16
    gemm_bf16_kernel<<<dim3(Fz / 128, BTz / 128), 256, 0, stream>>>(
        h2b, Cz, W1T, Cz, Cz, b1, nullptr, a1b, Fz, GEMM_RELU,
        nullptr, nullptr, nullptr);

    // FFN2: M=4096, N=1024, K=4096 -> f32 final output
    gemm_bf16_kernel<<<dim3(Cz / 128, BTz / 128), 256, 0, stream>>>(
        a1b, Fz, W2T, Fz, Fz, b2, out, nullptr, Cz, 0,
        nullptr, nullptr, nullptr);

    (void)in_sizes; (void)n_in; (void)out_size; (void)ws_size;
}
